// Attention_56143812493483
// MI455X (gfx1250) — matrix operand order repaired and checked by reading, not compiled, not run on hardware
//
#include <hip/hip_runtime.h>
#include <hip/hip_bf16.h>

// ---------------------------------------------------------------------------
// Problem sizes (fixed by the reference)
#define H   512
#define B2H 1024   // 2*H
#define BSZ 32     // batch
#define S   2048   // src len
#define M_TOTAL (S * BSZ)          // 65536 GEMM rows
#define NT_TOTAL (H / 16)          // 32 N-tiles
#define KT_TOTAL (B2H / 32)        // 32 K-steps of 32
#define MTILE 64                   // rows per workgroup (4 strips of 16)
#define GEMM_THREADS 512           // 16 waves; each wave owns 2 N-tiles
#define LDSTR (B2H + 8)            // padded LDS row stride (halves), keeps 16B align

typedef __attribute__((ext_vector_type(16))) __bf16  v16bf;
typedef __attribute__((ext_vector_type(8)))  __bf16  v8bf;
typedef __attribute__((ext_vector_type(8)))  float   v8f;
typedef __attribute__((ext_vector_type(4)))  float   v4f;
typedef __attribute__((ext_vector_type(4)))  unsigned short ushort4v;

__device__ inline unsigned short f2bf(float f) {
    unsigned u = __float_as_uint(f);
    u += 0x7FFFu + ((u >> 16) & 1u);   // round-to-nearest-even
    return (unsigned short)(u >> 16);
}

// CDNA5 has a native V_TANH_F32 transcendental op; use it when the toolchain
// exposes the builtin, else fall back to libm tanhf.
__device__ inline float fast_tanhf(float x) {
#if __has_builtin(__builtin_amdgcn_tanhf)
    return __builtin_amdgcn_tanhf(x);
#else
    return tanhf(x);
#endif
}

// ---------------------------------------------------------------------------
// Kernel 1: hb[b, n] = bias[n] + hidden[b,:] . attn_w[n, 0:1024]   (f32 exact)
__global__ void prep_hb_kernel(const float* __restrict__ hidden,
                               const float* __restrict__ attn_w,
                               const float* __restrict__ bias,
                               float* __restrict__ hb) {
    int t = blockIdx.x * 256 + threadIdx.x;      // 0 .. 16383
    int b = t >> 9;
    int n = t & (H - 1);
    const float4* hv = reinterpret_cast<const float4*>(hidden + (size_t)b * B2H);
    const float4* wv = reinterpret_cast<const float4*>(attn_w + (size_t)n * (4 * H));
    float s = bias[n];
    #pragma unroll 4
    for (int i = 0; i < B2H / 4; ++i) {
        float4 h = hv[i], w = wv[i];
        s += h.x * w.x + h.y * w.y + h.z * w.z + h.w * w.w;
    }
    hb[t] = s;
}

// ---------------------------------------------------------------------------
// Kernel 2: pack We = attn_w[:, 1024:2048] into bf16 WMMA B-fragment order.
// Fragment (kt, nt): 32 lanes x 16 halves contiguous.
//   lane < 16 : column n = nt*16+lane, K = kt*32 + (j<8 ? j : 16+j-8)      (j = 0..15)
//   lane >= 16: column n = nt*16+lane-16, K = kt*32 + (j<8 ? 8+j : 24+j-8)
__global__ void pack_B_kernel(const float* __restrict__ attn_w,
                              unsigned short* __restrict__ Bp) {
    int o = blockIdx.x * 256 + threadIdx.x;      // 0 .. 524287
    int j    = o & 15;
    int lane = (o >> 4) & 31;
    int nt   = (o >> 9) & 31;
    int kt   = o >> 14;
    int n = (nt << 4) + (lane & 15);
    int k = (kt << 5) + ((j >> 3) << 4) + ((lane >> 4) << 3) + (j & 7);
    Bp[o] = f2bf(attn_w[(size_t)n * (4 * H) + B2H + k]);
}

// ---------------------------------------------------------------------------
// Kernel 3: per-block 64-row M strip: bf16 WMMA GEMM over K=1024, N=512.
// 16 waves; each wave owns 2 N-tiles across all 4 row-strips (8 accumulator
// tiles = 64 VGPRs, no spilling) while every B fragment still feeds 4 WMMAs
// (4x less L2 traffic than M=16 tiling).
// Epilogue: tanh(acc + hb)*v reduced over N -> raw score into d_out[b*S+s].
__global__ void gemm_tanh_reduce_kernel(const float* __restrict__ eo,
                                        const unsigned short* __restrict__ Bp,
                                        const float* __restrict__ hb,
                                        const float* __restrict__ vvec,
                                        float* __restrict__ attRaw) {
    __shared__ __align__(16) unsigned short smA[MTILE * LDSTR];  // ~132 KB
    __shared__ float smAcc[MTILE];

    const int tid  = threadIdx.x;
    const int lane = tid & 31;
    const int wave = tid >> 5;               // 0..15
    const int m_base = blockIdx.x * MTILE;

    // ---- stage A strip (64 rows x 1024 K, f32 -> bf16) into LDS ----
    // eo is a read-once 268 MB stream: non-temporal loads keep the L2 for
    // the reused Bp/hb/v working set.
    const v4f* src = reinterpret_cast<const v4f*>(eo + (size_t)m_base * B2H);
    for (int i = 0; i < MTILE * (B2H / 4) / GEMM_THREADS; ++i) {   // 32 iters
        int idx = tid + i * GEMM_THREADS;    // float4 index, 0..16383
        v4f f = __builtin_nontemporal_load(&src[idx]);
        int e   = idx << 2;
        int row = e >> 10;
        int col = e & (B2H - 1);
        ushort4v p;
        p[0] = f2bf(f[0]); p[1] = f2bf(f[1]); p[2] = f2bf(f[2]); p[3] = f2bf(f[3]);
        *reinterpret_cast<ushort4v*>(&smA[row * LDSTR + col]) = p;
    }
    if (tid < MTILE) smAcc[tid] = 0.0f;
    __syncthreads();

    // ---- WMMA K loop ----
    const int row = lane & 15;
    const int sel = lane >> 4;               // half-wave select
    const int ntg0 = wave * 2;               // first of this wave's 2 N-tiles

    v8f acc[4][2];                           // [row-strip][n-tile]
    v8f zero = {0.f, 0.f, 0.f, 0.f, 0.f, 0.f, 0.f, 0.f};
    #pragma unroll
    for (int g = 0; g < 4; ++g)
        #pragma unroll
        for (int t = 0; t < 2; ++t) acc[g][t] = zero;

    for (int kk = 0; kk < KT_TOTAL; ++kk) {
        int k0 = kk << 5;
        // A fragments for the 4 row-strips (ISA 16-bit A 16x32 layout)
        v16bf a[4];
        #pragma unroll
        for (int g = 0; g < 4; ++g) {
            int r = (g << 4) + row;
            v8bf alo = *reinterpret_cast<const v8bf*>(&smA[r * LDSTR + k0 + sel * 8]);
            v8bf ahi = *reinterpret_cast<const v8bf*>(&smA[r * LDSTR + k0 + 16 + sel * 8]);
            a[g] = __builtin_shufflevector(alo, ahi,
                     0, 1, 2, 3, 4, 5, 6, 7, 8, 9, 10, 11, 12, 13, 14, 15);
        }
        const v16bf* bp = reinterpret_cast<const v16bf*>(Bp)
                        + ((size_t)(kk * NT_TOTAL + ntg0) * 32 + lane);
        #pragma unroll
        for (int t = 0; t < 2; ++t) {
            v16bf bfrag = bp[(size_t)t * 32];
            #pragma unroll
            for (int g = 0; g < 4; ++g) {
                acc[g][t] = __builtin_amdgcn_wmma_f32_16x16x32_bf16(
                                false, a[g], false, bfrag, (short)0, acc[g][t],
                                false, false);
            }
        }
    }

    // ---- epilogue: tanh(acc + hb)*v, reduce over N ----
    float rs[4][8];
    #pragma unroll
    for (int g = 0; g < 4; ++g)
        #pragma unroll
        for (int r = 0; r < 8; ++r) rs[g][r] = 0.0f;

    #pragma unroll
    for (int t = 0; t < 2; ++t) {
        int n = ((ntg0 + t) << 4) + row;     // C-tile column for this lane
        float vv = vvec[n];
        #pragma unroll
        for (int g = 0; g < 4; ++g) {
            #pragma unroll
            for (int r = 0; r < 8; ++r) {
                // m_base is a multiple of 64, so b depends only on local row
                int b = ((g << 4) + r + (sel << 3)) & (BSZ - 1);
                float e = acc[g][t][r] + hb[b * H + n];
                rs[g][r] += fast_tanhf(e) * vv;
            }
        }
    }
    // butterfly-sum across the 16 lanes sharing each M row
    #pragma unroll
    for (int msk = 1; msk < 16; msk <<= 1) {
        #pragma unroll
        for (int g = 0; g < 4; ++g)
            #pragma unroll
            for (int r = 0; r < 8; ++r) rs[g][r] += __shfl_xor(rs[g][r], msk, 32);
    }
    if ((lane & 15) == 0) {
        #pragma unroll
        for (int g = 0; g < 4; ++g)
            #pragma unroll
            for (int r = 0; r < 8; ++r)
                atomicAdd(&smAcc[(g << 4) + (sel << 3) + r], rs[g][r]);
    }
    __syncthreads();

    if (tid < MTILE) {
        int m = m_base + tid;
        int b = m & (BSZ - 1);
        int s = m >> 5;
        attRaw[(size_t)b * S + s] = smAcc[tid];
    }
}

// ---------------------------------------------------------------------------
// Kernel 4: masked softmax over S per batch row, in place on d_out.
__global__ void softmax_mask_kernel(float* __restrict__ att,
                                    const int* __restrict__ mask) {
    __shared__ float red[256];
    const int b = blockIdx.x;
    const int tid = threadIdx.x;

    float x[8];
    float mx = -3.4e38f;
    #pragma unroll
    for (int i = 0; i < 8; ++i) {
        int s = tid + i * 256;
        float r = att[(size_t)b * S + s];
        if (mask[(size_t)b * S + s] == 0) r = -1e10f;
        x[i] = r;
        mx = fmaxf(mx, r);
    }
    red[tid] = mx;
    __syncthreads();
    for (int o = 128; o > 0; o >>= 1) {
        if (tid < o) red[tid] = fmaxf(red[tid], red[tid + o]);
        __syncthreads();
    }
    mx = red[0];
    __syncthreads();

    float sum = 0.0f;
    #pragma unroll
    for (int i = 0; i < 8; ++i) { x[i] = __expf(x[i] - mx); sum += x[i]; }
    red[tid] = sum;
    __syncthreads();
    for (int o = 128; o > 0; o >>= 1) {
        if (tid < o) red[tid] += red[tid + o];
        __syncthreads();
    }
    float inv = 1.0f / red[0];
    #pragma unroll
    for (int i = 0; i < 8; ++i) att[(size_t)b * S + tid + i * 256] = x[i] * inv;
}

// ---------------------------------------------------------------------------
extern "C" void kernel_launch(void* const* d_in, const int* in_sizes, int n_in,
                              void* d_out, int out_size, void* d_ws, size_t ws_size,
                              hipStream_t stream) {
    const float* hidden = (const float*)d_in[0];   // [B, 2H]
    const float* eo     = (const float*)d_in[1];   // [S, B, 2H]
    const int*   mask   = (const int*)d_in[2];     // [B, S]
    const float* attn_w = (const float*)d_in[3];   // [H, 4H]
    const float* attn_b = (const float*)d_in[4];   // [H]
    const float* v      = (const float*)d_in[5];   // [H]
    float* out = (float*)d_out;                    // [B, S]

    // workspace layout: hb (64 KB f32) | Bpack (1 MB bf16)
    float* hb = (float*)d_ws;
    unsigned short* Bp = (unsigned short*)((char*)d_ws + (size_t)BSZ * H * sizeof(float));

    prep_hb_kernel<<<(BSZ * H) / 256, 256, 0, stream>>>(hidden, attn_w, attn_b, hb);
    pack_B_kernel<<<(B2H * H) / 256, 256, 0, stream>>>(attn_w, Bp);
    gemm_tanh_reduce_kernel<<<M_TOTAL / MTILE, GEMM_THREADS, 0, stream>>>(eo, Bp, hb, v, out);
    softmax_mask_kernel<<<BSZ, 256, 0, stream>>>(out, mask);
}
